// TransformerBlock_72593537237093
// MI455X (gfx1250) — matrix operand order repaired and checked by reading, not compiled, not run on hardware
//
#include <hip/hip_runtime.h>
#include <hip/hip_bf16.h>
#include <cstdint>

typedef __bf16 bf16_t;
typedef __attribute__((ext_vector_type(8)))  __bf16 v8bf;
typedef __attribute__((ext_vector_type(16))) __bf16 v16bf;
typedef __attribute__((ext_vector_type(8)))  float  v8f;
typedef __attribute__((ext_vector_type(4))) unsigned int v4u;
typedef __attribute__((ext_vector_type(8))) int v8i;
typedef __attribute__((ext_vector_type(4))) int v4i;

constexpr int BB  = 2;
constexpr int SS  = 2048;
constexpr int DD  = 2048;
constexpr int HH  = 16;
constexpr int HDM = 128;
constexpr int NTOK = BB * SS;   // 4096
constexpr int D3 = 3 * DD;      // 6144
constexpr int D4 = 4 * DD;      // 8192

// Combine two 16-byte chunks into one 16-element bf16 WMMA fragment register set.
static __device__ __forceinline__ v16bf make_frag(const bf16_t* p0, const bf16_t* p1) {
  v8bf lo = *(const v8bf*)p0;
  v8bf hi = *(const v8bf*)p1;
  return __builtin_shufflevector(lo, hi, 0,1,2,3,4,5,6,7,8,9,10,11,12,13,14,15);
}

// ---------------- Tensor Data Mover: 2D bf16 tile -> LDS (ISA ch.8 descriptor) ----
// Loads a tile of tile_d1 rows x 32 bf16 cols from a row-major [d1, stride] tensor.
// TDM pad feature inserts 4 DWORDs (8 bf16) after each 16-DWORD (32 bf16) row,
// reproducing the 40-element padded LDS row stride used by the fragment loaders.
static __device__ __forceinline__ void tdm_load_tile_2d(
    const bf16_t* gptr, unsigned lds_off, int tensor_d0, int tensor_d1,
    int stride_elems) {
  const unsigned long long ga = (unsigned long long)(uintptr_t)gptr;
  v4u g0;
  g0[0] = 1u;                                    // count=1 valid descriptor
  g0[1] = lds_off;                               // lds_addr (bytes)
  g0[2] = (unsigned)(ga & 0xFFFFFFFFu);          // global_addr[31:0]
  g0[3] = (unsigned)((ga >> 32) & 0x1FFFFFFu)    // global_addr[56:32]
        | (2u << 30);                            // type = 2 ("image")
  v8i g1;
  g1[0] = (int)((1u << 16)                       // data_size = 1 -> 2 bytes
              | (1u << 20)                       // pad_enable
              | (3u << 22)                       // pad_interval: 16 DWORDs
              | (3u << 25));                     // pad_amount:   4 DWORDs
  g1[1] = (int)(((unsigned)tensor_d0 & 0xFFFFu) << 16);            // tensor_dim0[15:0]
  g1[2] = (int)((((unsigned)tensor_d0 >> 16) & 0xFFFFu)            // tensor_dim0[31:16]
              | (((unsigned)tensor_d1 & 0xFFFFu) << 16));          // tensor_dim1[15:0]
  g1[3] = (int)((((unsigned)tensor_d1 >> 16) & 0xFFFFu)            // tensor_dim1[31:16]
              | (32u << 16));                                      // tile_dim0 = 32
  g1[4] = (int)(128u);                                             // tile_dim1=128, tile_dim2=0
  g1[5] = (int)stride_elems;                                       // tensor_dim0_stride[31:0]
  g1[6] = 0;                                                       // stride hi / dim1_stride
  g1[7] = 0;
  const v4i z4 = {0, 0, 0, 0};
#if defined(__clang_major__) && __clang_major__ >= 23
  const v8i z8 = {0, 0, 0, 0, 0, 0, 0, 0};
  __builtin_amdgcn_tensor_load_to_lds(g0, g1, z4, z4, z8, 0);
#else
  __builtin_amdgcn_tensor_load_to_lds(g0, g1, z4, z4, 0);
#endif
}

// ---------------- fp32 -> bf16 convert ----------------
__global__ __launch_bounds__(256) void cvt_f32_bf16(const float* __restrict__ in,
                                                    bf16_t* __restrict__ out, int n) {
  int i = (blockIdx.x * 256 + threadIdx.x) * 4;
  if (i + 3 < n) {
    float4 f = *(const float4*)(in + i);
    out[i + 0] = (bf16_t)f.x;
    out[i + 1] = (bf16_t)f.y;
    out[i + 2] = (bf16_t)f.z;
    out[i + 3] = (bf16_t)f.w;
  }
}

// ---------------- RMSNorm (fp32 in, bf16 out), one block per token ----------------
__global__ __launch_bounds__(256) void rmsnorm_to_bf16(const float* __restrict__ x,
                                                       const float* __restrict__ g,
                                                       bf16_t* __restrict__ out) {
  __shared__ float red[8];
  const int row  = blockIdx.x;
  const int base = threadIdx.x * 8;   // D = 2048 = 256 threads * 8
  const float* xr = x + (size_t)row * DD;
  float4 a = *(const float4*)(xr + base);
  float4 b = *(const float4*)(xr + base + 4);
  float s = a.x*a.x + a.y*a.y + a.z*a.z + a.w*a.w
          + b.x*b.x + b.y*b.y + b.z*b.z + b.w*b.w;
  for (int off = 16; off > 0; off >>= 1) s += __shfl_xor(s, off);   // wave32 reduce
  if ((threadIdx.x & 31) == 0) red[threadIdx.x >> 5] = s;
  __syncthreads();
  float tot = red[0]+red[1]+red[2]+red[3]+red[4]+red[5]+red[6]+red[7];
  float rinv = rsqrtf(tot * (1.0f / DD) + 1e-6f);
  float4 ga = *(const float4*)(g + base);
  float4 gb = *(const float4*)(g + base + 4);
  bf16_t* o = out + (size_t)row * DD + base;
  o[0] = (bf16_t)(a.x * rinv * ga.x);
  o[1] = (bf16_t)(a.y * rinv * ga.y);
  o[2] = (bf16_t)(a.z * rinv * ga.z);
  o[3] = (bf16_t)(a.w * rinv * ga.w);
  o[4] = (bf16_t)(b.x * rinv * gb.x);
  o[5] = (bf16_t)(b.y * rinv * gb.y);
  o[6] = (bf16_t)(b.z * rinv * gb.z);
  o[7] = (bf16_t)(b.w * rinv * gb.w);
}

// ---------------- Tiled WMMA GEMM with TDM-staged, double-buffered LDS ------------
// C[M,N] = A[M,K] * W[N,K]^T
// EPI: 0 = plain fp32 out, 1 = fp32 out + residual add, 2 = tanh-GELU -> bf16 out
template <int EPI>
__global__ __launch_bounds__(256) void gemm_bf16_wmma(
    const bf16_t* __restrict__ A, const bf16_t* __restrict__ W,
    const float* __restrict__ resid, float* __restrict__ Cf,
    bf16_t* __restrict__ Cb, int M, int N, int K) {
  constexpr int LDT = 40;                  // 32 + 8 bf16 pad (TDM pad feature)
  __shared__ bf16_t As[2][128 * LDT];
  __shared__ bf16_t Bs[2][128 * LDT];
  const int tid  = threadIdx.x;
  const int lane = tid & 31;
  const int wid  = tid >> 5;               // 8 waves
  const int l16  = lane & 15;
  const int lh   = lane >> 4;
  const int wrow = (wid & 1) * 64;         // 2x4 wave grid -> 64x32 per wave
  const int wcol = (wid >> 1) * 32;
  const int m0 = blockIdx.y * 128;
  const int n0 = blockIdx.x * 128;
  // ISA 7.12.2 fragment chunk offsets:
  const int cA0 = lh * 8, cA1 = 16 + lh * 8;     // A: K 0-7/16-23 | 8-15/24-31
  const int cB0 = lh * 8, cB1 = 16 + lh * 8;     // B: K 0-7/16-23 | 8-15/24-31

  const unsigned asOff0 = (unsigned)(uintptr_t)&As[0][0];
  const unsigned asOff1 = (unsigned)(uintptr_t)&As[1][0];
  const unsigned bsOff0 = (unsigned)(uintptr_t)&Bs[0][0];
  const unsigned bsOff1 = (unsigned)(uintptr_t)&Bs[1][0];

  v8f acc[4][2] = {};
  const int nk = K / 32;

  // Prologue: wave 0 kicks off tile 0 via the Tensor Data Mover.
  if (wid == 0) {
    tdm_load_tile_2d(A + (size_t)m0 * K, asOff0, K, M, K);
    tdm_load_tile_2d(W + (size_t)n0 * K, bsOff0, K, N, K);
  }

  for (int kt = 0; kt < nk; ++kt) {
    const int cur = kt & 1;
    if (wid == 0) {
      if (kt + 1 < nk) {
        const int k1 = (kt + 1) * 32;
        const unsigned an = (cur ? asOff0 : asOff1);
        const unsigned bn = (cur ? bsOff0 : bsOff1);
        tdm_load_tile_2d(A + (size_t)m0 * K + k1, an, K, M, K);
        tdm_load_tile_2d(W + (size_t)n0 * K + k1, bn, K, N, K);
        __builtin_amdgcn_s_wait_tensorcnt(2);   // in-order: tile kt's pair landed
      } else {
        __builtin_amdgcn_s_wait_tensorcnt(0);
      }
    }
    __syncthreads();

    const bf16_t* at = &As[cur][0];
    const bf16_t* bt = &Bs[cur][0];
    v16bf af[4], bq[2];
    for (int i = 0; i < 4; ++i) {
      const bf16_t* p = at + (wrow + i * 16 + l16) * LDT;
      af[i] = make_frag(p + cA0, p + cA1);
    }
    for (int j = 0; j < 2; ++j) {
      const bf16_t* p = bt + (wcol + j * 16 + l16) * LDT;
      bq[j] = make_frag(p + cB0, p + cB1);
    }
    for (int i = 0; i < 4; ++i)
      for (int j = 0; j < 2; ++j)
        acc[i][j] = __builtin_amdgcn_wmma_f32_16x16x32_bf16(
            false, af[i], false, bq[j], (short)0, acc[i][j], false, false);

    __syncthreads();   // reads done; next TDM may overwrite this buffer
  }

  for (int i = 0; i < 4; ++i) {
    const int rowb = m0 + wrow + i * 16 + lh * 8;    // C layout: lane half picks M 0-7/8-15
    for (int j = 0; j < 2; ++j) {
      const int col = n0 + wcol + j * 16 + l16;
      for (int r = 0; r < 8; ++r) {
        const size_t idx = (size_t)(rowb + r) * N + col;
        float v = acc[i][j][r];
        if constexpr (EPI == 0) {
          Cf[idx] = v;
        } else if constexpr (EPI == 1) {
          Cf[idx] = resid[idx] + v;
        } else {
          const float t = tanhf(0.7978845608028654f * (v + 0.044715f * v * v * v));
          Cb[idx] = (bf16_t)(0.5f * v * (1.0f + t));
        }
      }
    }
  }
}

// ---------------- RoPE + split heads: qkv fp32 [tok, 3, H, HD] -> bf16 [B,H,S,HD] ----------------
__global__ __launch_bounds__(256) void rope_split(
    const float* __restrict__ qkv, const float* __restrict__ cosb,
    const float* __restrict__ sinb, bf16_t* __restrict__ qh,
    bf16_t* __restrict__ kh, bf16_t* __restrict__ vh) {
  const int idx = blockIdx.x * 256 + threadIdx.x;   // exactly NTOK*D threads
  const int d = idx & (HDM - 1);
  const int h = (idx >> 7) & (HH - 1);
  const int t = idx >> 11;                          // b*S + s
  const int s = t & (SS - 1);
  const int b = t >> 11;
  const size_t qoff = (size_t)t * D3 + h * HDM;
  const float c  = cosb[s * HDM + d];
  const float sn = sinb[s * HDM + d];
  const int   dp  = (d < HDM / 2) ? d + HDM / 2 : d - HDM / 2;
  const float sgn = (d < HDM / 2) ? -1.f : 1.f;
  const float q  = qkv[qoff + d],      qp = qkv[qoff + dp];
  const float k  = qkv[qoff + DD + d], kp = qkv[qoff + DD + dp];
  const float vv = qkv[qoff + 2 * DD + d];
  const size_t oo = (((size_t)b * HH + h) * SS + s) * HDM + d;
  qh[oo] = (bf16_t)(q * c + sgn * qp * sn);
  kh[oo] = (bf16_t)(k * c + sgn * kp * sn);
  vh[oo] = (bf16_t)vv;
}

// ---------------- Flash-style causal attention (bf16 WMMA, fp32 softmax) ----------------
constexpr int KP = 136;   // K tile row stride (128 + 8)
constexpr int VP = 72;    // Vt / P row stride (64 + 8)

__global__ __launch_bounds__(128) void attn_fwd(
    const bf16_t* __restrict__ qg, const bf16_t* __restrict__ kg,
    const bf16_t* __restrict__ vg, bf16_t* __restrict__ ob) {
  __shared__ bf16_t Ks[64 * KP];        // K tile, row-major [key][d]
  __shared__ bf16_t Vt[128 * VP];       // V tile transposed [d][key]
  __shared__ bf16_t Ps[4][16 * VP];     // per-wave P tile [qrow][key]
  const int tid = threadIdx.x, lane = tid & 31, wid = tid >> 5;
  const int l16 = lane & 15, lh = lane >> 4;
  const int cA0 = lh * 8,  cA1 = 16 + lh * 8;
  const int cB0 = lh * 8,  cB1 = 16 + lh * 8;
  const int b = blockIdx.z, h = blockIdx.y, qt = blockIdx.x;
  const size_t hb = ((size_t)b * HH + h) * SS;
  const bf16_t* qp = qg + hb * HDM;
  const bf16_t* kp = kg + hb * HDM;
  const bf16_t* vp = vg + hb * HDM;
  const int q0 = qt * 64;
  const int qrow0 = q0 + wid * 16 + lh * 8;

  // Q fragments (16 q-rows x 128 d per wave) loaded once, kept in registers.
  v16bf qf[4];
  {
    const bf16_t* qr = qp + (size_t)(q0 + wid * 16 + l16) * HDM;
    for (int kk = 0; kk < 4; ++kk)
      qf[kk] = make_frag(qr + kk * 32 + cA0, qr + kk * 32 + cA1);
  }

  v8f of[8] = {};
  float mloc[8], lsum[8];
  for (int r = 0; r < 8; ++r) { mloc[r] = -1e30f; lsum[r] = 0.f; }
  const float scale = 0.08838834764831845f;   // 1/sqrt(128)

  for (int kt = 0; kt <= qt; ++kt) {
    const int k0 = kt * 64;
    __syncthreads();
    {   // stage K row-major and V transposed
      int r = tid >> 4;
      const int cc = (tid & 15) * 8;
      for (int p = 0; p < 8; ++p, r += 8) {
        *(v8bf*)&Ks[r * KP + cc] = *(const v8bf*)(kp + (size_t)(k0 + r) * HDM + cc);
        v8bf vv = *(const v8bf*)(vp + (size_t)(k0 + r) * HDM + cc);
        for (int e = 0; e < 8; ++e) Vt[(cc + e) * VP + r] = vv[e];
      }
    }
    __syncthreads();

    // S = Q * K^T  (16x64 per wave)
    v8f sc[4] = {};
    for (int fn = 0; fn < 4; ++fn) {
      const bf16_t* krow = &Ks[(fn * 16 + l16) * KP];
      for (int kk = 0; kk < 4; ++kk) {
        v16bf kf = make_frag(krow + kk * 32 + cB0, krow + kk * 32 + cB1);
        sc[fn] = __builtin_amdgcn_wmma_f32_16x16x32_bf16(
            false, qf[kk], false, kf, (short)0, sc[fn], false, false);
      }
    }

    // scale + causal mask + online softmax
    float tmax[8];
    for (int r = 0; r < 8; ++r) tmax[r] = -1e30f;
    for (int fn = 0; fn < 4; ++fn) {
      const int kidx = k0 + fn * 16 + l16;
      for (int r = 0; r < 8; ++r) {
        float v = sc[fn][r] * scale;
        v = (kidx > qrow0 + r) ? -1e30f : v;
        sc[fn][r] = v;
        tmax[r] = fmaxf(tmax[r], v);
      }
    }
    for (int off = 1; off < 16; off <<= 1)
      for (int r = 0; r < 8; ++r)
        tmax[r] = fmaxf(tmax[r], __shfl_xor(tmax[r], off));
    float corr[8];
    for (int r = 0; r < 8; ++r) {
      const float mn = fmaxf(mloc[r], tmax[r]);
      corr[r] = __expf(mloc[r] - mn);
      mloc[r] = mn;
    }
    float rsum[8] = {};
    for (int fn = 0; fn < 4; ++fn)
      for (int r = 0; r < 8; ++r) {
        const float p = __expf(sc[fn][r] - mloc[r]);
        sc[fn][r] = p;
        rsum[r] += p;
      }
    for (int off = 1; off < 16; off <<= 1)
      for (int r = 0; r < 8; ++r) rsum[r] += __shfl_xor(rsum[r], off);
    for (int r = 0; r < 8; ++r) lsum[r] = lsum[r] * corr[r] + rsum[r];
    for (int j = 0; j < 8; ++j)
      for (int r = 0; r < 8; ++r) of[j][r] *= corr[r];

    // P (C-layout) -> per-wave LDS -> A-layout fragments
    for (int fn = 0; fn < 4; ++fn)
      for (int r = 0; r < 8; ++r)
        Ps[wid][(lh * 8 + r) * VP + fn * 16 + l16] = (bf16_t)sc[fn][r];
    asm volatile("s_wait_dscnt 0" ::: "memory");
    v16bf pf[2];
    {
      const bf16_t* pr = &Ps[wid][l16 * VP];
      for (int t = 0; t < 2; ++t)
        pf[t] = make_frag(pr + t * 32 + cA0, pr + t * 32 + cA1);
    }

    // O += P * V  (16x128 per wave)
    for (int j = 0; j < 8; ++j) {
      const bf16_t* vrow = &Vt[(j * 16 + l16) * VP];
      for (int t = 0; t < 2; ++t) {
        v16bf vf = make_frag(vrow + t * 32 + cB0, vrow + t * 32 + cB1);
        of[j] = __builtin_amdgcn_wmma_f32_16x16x32_bf16(
            false, pf[t], false, vf, (short)0, of[j], false, false);
      }
    }
  }

  float inv[8];
  for (int r = 0; r < 8; ++r) inv[r] = 1.0f / lsum[r];
  for (int j = 0; j < 8; ++j) {
    const int d = j * 16 + l16;
    for (int r = 0; r < 8; ++r) {
      const int q = qrow0 + r;
      ob[((size_t)(b * SS) + q) * DD + h * HDM + d] = (bf16_t)(of[j][r] * inv[r]);
    }
  }
}

// ---------------- host launcher ----------------
extern "C" void kernel_launch(void* const* d_in, const int* in_sizes, int n_in,
                              void* d_out, int out_size, void* d_ws, size_t ws_size,
                              hipStream_t stream) {
  (void)in_sizes; (void)n_in; (void)out_size; (void)ws_size;
  const float* x    = (const float*)d_in[0];
  const float* cosb = (const float*)d_in[1];
  const float* sinb = (const float*)d_in[2];
  const float* g1   = (const float*)d_in[3];
  const float* g2   = (const float*)d_in[4];
  const float* wqkv = (const float*)d_in[5];
  const float* wout = (const float*)d_in[6];
  const float* wfc1 = (const float*)d_in[7];
  const float* wfc2 = (const float*)d_in[8];

  char* wsp = (char*)d_ws;
  auto alloc = [&](size_t bytes) -> char* {
    char* p = wsp;
    wsp += (bytes + 255) & ~(size_t)255;
    return p;
  };
  bf16_t* wqkv_b = (bf16_t*)alloc((size_t)D3 * DD * 2);
  bf16_t* wout_b = (bf16_t*)alloc((size_t)DD * DD * 2);
  bf16_t* wfc1_b = (bf16_t*)alloc((size_t)D4 * DD * 2);
  bf16_t* wfc2_b = (bf16_t*)alloc((size_t)DD * D4 * 2);
  bf16_t* h_b    = (bf16_t*)alloc((size_t)NTOK * DD * 2);
  bf16_t* o_b    = (bf16_t*)alloc((size_t)NTOK * DD * 2);
  bf16_t* f1_b   = (bf16_t*)alloc((size_t)NTOK * D4 * 2);
  float*  qkv    = (float*) alloc((size_t)NTOK * D3 * 4);
  bf16_t* qh     = (bf16_t*)alloc((size_t)NTOK * DD * 2);
  bf16_t* kh     = (bf16_t*)alloc((size_t)NTOK * DD * 2);
  bf16_t* vh     = (bf16_t*)alloc((size_t)NTOK * DD * 2);
  float*  x2     = (float*) alloc((size_t)NTOK * DD * 4);

  auto cvt = [&](const float* src, bf16_t* dst, size_t n) {
    cvt_f32_bf16<<<dim3((unsigned)((n / 4 + 255) / 256)), dim3(256), 0, stream>>>(src, dst, (int)n);
  };
  cvt(wqkv, wqkv_b, (size_t)D3 * DD);
  cvt(wout, wout_b, (size_t)DD * DD);
  cvt(wfc1, wfc1_b, (size_t)D4 * DD);
  cvt(wfc2, wfc2_b, (size_t)DD * D4);

  // h = rmsnorm(x, g1)
  rmsnorm_to_bf16<<<NTOK, 256, 0, stream>>>(x, g1, h_b);
  // qkv = h @ w_qkv^T
  gemm_bf16_wmma<0><<<dim3(D3 / 128, NTOK / 128), 256, 0, stream>>>(
      h_b, wqkv_b, nullptr, qkv, nullptr, NTOK, D3, DD);
  // rope + head split
  rope_split<<<(NTOK * DD) / 256, 256, 0, stream>>>(qkv, cosb, sinb, qh, kh, vh);
  // causal flash attention
  attn_fwd<<<dim3(SS / 64, HH, BB), 128, 0, stream>>>(qh, kh, vh, o_b);
  // x2 = x + o @ w_out^T
  gemm_bf16_wmma<1><<<dim3(DD / 128, NTOK / 128), 256, 0, stream>>>(
      o_b, wout_b, x, x2, nullptr, NTOK, DD, DD);
  // h = rmsnorm(x2, g2)
  rmsnorm_to_bf16<<<NTOK, 256, 0, stream>>>(x2, g2, h_b);
  // f1 = gelu(h @ w_fc1^T)
  gemm_bf16_wmma<2><<<dim3(D4 / 128, NTOK / 128), 256, 0, stream>>>(
      h_b, wfc1_b, nullptr, nullptr, f1_b, NTOK, D4, DD);
  // out = x2 + f1 @ w_fc2^T
  gemm_bf16_wmma<1><<<dim3(DD / 128, NTOK / 128), 256, 0, stream>>>(
      f1_b, wfc2_b, x2, (float*)d_out, nullptr, NTOK, DD, D4);
}
